// RNN_5179730559366
// MI455X (gfx1250) — matrix operand order repaired and checked by reading, not compiled, not run on hardware
//
#include <hip/hip_runtime.h>

typedef _Float16 f16;
typedef _Float16 v2h  __attribute__((ext_vector_type(2)));
typedef _Float16 v8h  __attribute__((ext_vector_type(8)));
typedef _Float16 v16h __attribute__((ext_vector_type(16)));
typedef float    v8f  __attribute__((ext_vector_type(8)));

#define T_STEPS 256
#define BATCH   64
#define DIN     2048
#define HID     1024
#define N3      3072          // 3 gates * HID
#define NCLS    1000
#define NROWS   (BATCH * T_STEPS)   // 16384

// ---------------------------------------------------------------------------
// Build effective TT matrix: Weff[d][g*HID + h] (f16)
// W = sum_{r1,r2,r3} C0[g,i,p,r1] C1[g,r1,j,q,r2] C2[g,r2,k,u,r3] C3[g,r3,l,v]
// d = ((i*8+j)*8+k)*8+l ; h = ((p*4+q)*8+u)*8+v
// ---------------------------------------------------------------------------
__global__ __launch_bounds__(256) void build_weff_kernel(
    const float* __restrict__ c0, const float* __restrict__ c1,
    const float* __restrict__ c2, const float* __restrict__ c3,
    f16* __restrict__ Weff)
{
  int idx = blockIdx.x * 256 + threadIdx.x;       // over 3*2048*1024
  if (idx >= 3 * DIN * HID) return;
  int g   = idx / (DIN * HID);
  int rem = idx - g * (DIN * HID);
  int d = rem >> 10;
  int h = rem & 1023;
  int i = d >> 9, j = (d >> 6) & 7, k = (d >> 3) & 7, l = d & 7;
  int p = h >> 8, q = (h >> 6) & 3, u = (h >> 3) & 7, v = h & 7;

  float t3[4];
#pragma unroll
  for (int r2 = 0; r2 < 4; ++r2) {
    float s = 0.f;
#pragma unroll
    for (int r3 = 0; r3 < 4; ++r3)
      s += c2[g*1024 + r2*256 + k*32 + u*4 + r3] * c3[g*256 + r3*64 + l*8 + v];
    t3[r2] = s;
  }
  float acc = 0.f;
#pragma unroll
  for (int r1 = 0; r1 < 4; ++r1) {
    float s = 0.f;
#pragma unroll
    for (int r2 = 0; r2 < 4; ++r2)
      s += c1[g*512 + r1*128 + j*16 + q*4 + r2] * t3[r2];
    acc += c0[g*64 + i*16 + p*4 + r1] * s;
  }
  Weff[(size_t)d * N3 + g * HID + h] = (f16)acc;
}

__global__ __launch_bounds__(256) void f32_to_f16_kernel(
    const float* __restrict__ src, f16* __restrict__ dst, int n)
{
  int i = blockIdx.x * 256 + threadIdx.x;
  if (i < n) dst[i] = (f16)src[i];
}

__global__ __launch_bounds__(256) void zero_f32_kernel(float* __restrict__ p, int n)
{
  int i = blockIdx.x * 256 + threadIdx.x;
  if (i < n) p[i] = 0.f;
}

// ---------------------------------------------------------------------------
// Tiled WMMA GEMM with register-prefetch double buffering.
// C[M x N] = A[M x K](f32 -> f16 in LDS) * B[K x N](f16, LDS-transposed to [n][k])
// NT threads = WR x WC waves; wave tile (BM/WR) x (BN/WC); K-step 32.
// Requires BM | M, BN | N, 32 | K (all satisfied exactly; no bounds checks).
// ---------------------------------------------------------------------------
template <int NT, int BM, int BN, int WR, int WC, bool STORE_F16>
__global__ __launch_bounds__(NT) void gemm_wmma_kernel(
    const float* __restrict__ A, const f16* __restrict__ B,
    void* __restrict__ Cout, int M, int N, int K)
{
  constexpr int BK   = 32;
  constexpr int LDSK = BK + 8;            // 40 halves = 80B row stride (16B multiple)
  constexpr int WTM  = BM / WR;
  constexpr int WTN  = BN / WC;
  constexpr int MF   = WTM / 16;
  constexpr int NF   = WTN / 16;
  constexpr int APAIRS = (BM * BK) / (2 * NT);   // float2 pairs per thread (A)
  constexpr int BPAIRS = (BK * BN) / (2 * NT);   // k-pairs per thread (B)
  static_assert(WR * WC * 32 == NT, "wave count");
  static_assert(APAIRS >= 1 && BPAIRS >= 1, "staging divisibility");

  __shared__ __align__(16) f16 sA[2][BM][LDSK];   // [buf][m][k]
  __shared__ __align__(16) f16 sB[2][BN][LDSK];   // [buf][n][k] (transposed)

  const int tid   = threadIdx.x;
  const int lane  = tid & 31;
  const int wave  = tid >> 5;
  const int wr    = wave / WC;
  const int wc    = wave % WC;
  const int lmod  = lane & 15;
  const int lhalf = lane >> 4;

  const int rowBase = blockIdx.y * BM;
  const int colBase = blockIdx.x * BN;

  v8f acc[MF][NF];
#pragma unroll
  for (int a = 0; a < MF; ++a)
#pragma unroll
    for (int b = 0; b < NF; ++b)
#pragma unroll
      for (int e = 0; e < 8; ++e) acc[a][b][e] = 0.f;

  float2 aReg[APAIRS];
  f16    bRegL[BPAIRS], bRegH[BPAIRS];

  // ---- prologue: fetch first tile into registers ----
#pragma unroll
  for (int e = 0; e < APAIRS; ++e) {
    int p = e * NT + tid;
    int r = p >> 4;                        // BK/2 = 16 pairs per row
    int kp = p & 15;
    aReg[e] = *(const float2*)&A[(size_t)(rowBase + r) * K + 2 * kp];
  }
#pragma unroll
  for (int e = 0; e < BPAIRS; ++e) {
    int p  = e * NT + tid;
    int kp = p / BN;
    int nn = p - kp * BN;
    const f16* src = &B[(size_t)(2 * kp) * N + colBase + nn];
    bRegL[e] = src[0];
    bRegH[e] = src[N];
  }

  int buf = 0;
  for (int kt = 0; kt < K; kt += BK) {
    // ---- commit staged registers to LDS[buf] (packed b32 stores) ----
#pragma unroll
    for (int e = 0; e < APAIRS; ++e) {
      int p = e * NT + tid;
      int r = p >> 4;
      int kp = p & 15;
      v2h pa; pa[0] = (f16)aReg[e].x; pa[1] = (f16)aReg[e].y;
      *(v2h*)&sA[buf][r][2 * kp] = pa;
    }
#pragma unroll
    for (int e = 0; e < BPAIRS; ++e) {
      int p  = e * NT + tid;
      int kp = p / BN;
      int nn = p - kp * BN;
      v2h pb; pb[0] = bRegL[e]; pb[1] = bRegH[e];
      *(v2h*)&sB[buf][nn][2 * kp] = pb;
    }
    __syncthreads();

    // ---- issue next tile's global loads (overlaps with WMMA below) ----
    if (kt + BK < K) {
      const int ktn = kt + BK;
#pragma unroll
      for (int e = 0; e < APAIRS; ++e) {
        int p = e * NT + tid;
        int r = p >> 4;
        int kp = p & 15;
        aReg[e] = *(const float2*)&A[(size_t)(rowBase + r) * K + ktn + 2 * kp];
      }
#pragma unroll
      for (int e = 0; e < BPAIRS; ++e) {
        int p  = e * NT + tid;
        int kp = p / BN;
        int nn = p - kp * BN;
        const f16* src = &B[(size_t)(ktn + 2 * kp) * N + colBase + nn];
        bRegL[e] = src[0];
        bRegH[e] = src[N];
      }
    }

    // ---- fragment gather (ISA 7.12.2, wave32) + WMMA ----
    v16h afrag[MF];
#pragma unroll
    for (int a = 0; a < MF; ++a) {
      const f16* base = &sA[buf][wr * WTM + a * 16 + lmod][lhalf * 8];
      v8h lo = *(const v8h*)base;            // K = k0 .. k0+7
      v8h hi = *(const v8h*)(base + 16);     // K = k0+16 .. k0+23
      afrag[a] = __builtin_shufflevector(lo, hi, 0,1,2,3,4,5,6,7,8,9,10,11,12,13,14,15);
    }
    v16h bfrag[NF];
#pragma unroll
    for (int b = 0; b < NF; ++b) {
      const f16* base = &sB[buf][wc * WTN + b * 16 + lmod][lhalf * 8];
      v8h lo = *(const v8h*)base;            // K = k0 .. k0+7
      v8h hi = *(const v8h*)(base + 16);     // K = k0+16 .. k0+23
      bfrag[b] = __builtin_shufflevector(lo, hi, 0,1,2,3,4,5,6,7,8,9,10,11,12,13,14,15);
    }

#pragma unroll
    for (int a = 0; a < MF; ++a)
#pragma unroll
      for (int b = 0; b < NF; ++b)
        acc[a][b] = __builtin_amdgcn_wmma_f32_16x16x32_f16(
            false, afrag[a], false, bfrag[b], (short)0, acc[a][b], false, false);

    buf ^= 1;
  }

  // ---- store: elem j of lane L -> row = (L>>4)*8 + j, col = L&15 ----
#pragma unroll
  for (int a = 0; a < MF; ++a) {
#pragma unroll
    for (int b = 0; b < NF; ++b) {
      int col = colBase + wc * WTN + b * 16 + lmod;
#pragma unroll
      for (int jj = 0; jj < 8; ++jj) {
        int row = rowBase + wr * WTM + a * 16 + lhalf * 8 + jj;
        if constexpr (STORE_F16)
          ((f16*)Cout)[(size_t)row * N + col] = (f16)acc[a][b][jj];
        else
          ((float*)Cout)[(size_t)row * N + col] = acc[a][b][jj];
      }
    }
  }
}

// ---------------------------------------------------------------------------
// Fused GRU pointwise update (in-place on h), one thread per (b, h) element.
// ---------------------------------------------------------------------------
__global__ __launch_bounds__(256) void gru_pointwise_kernel(
    const f16* __restrict__ GI, const float* __restrict__ GH,
    const float* __restrict__ bi, const float* __restrict__ bh,
    float* __restrict__ h, int t)
{
  int idx = blockIdx.x * 256 + threadIdx.x;      // BATCH*HID = 65536
  int b  = idx >> 10;
  int hh = idx & 1023;

  const f16* gi = GI + (size_t)(b * T_STEPS + t) * N3;
  float gir = (float)gi[hh];
  float giz = (float)gi[HID + hh];
  float gin = (float)gi[2 * HID + hh];

  const float* gh = GH + (size_t)b * N3;
  float ghr = gh[hh], ghz = gh[HID + hh], ghn = gh[2 * HID + hh];

  float xr = gir + bi[hh]       + ghr + bh[hh];
  float xz = giz + bi[HID + hh] + ghz + bh[HID + hh];
  float r  = 1.f / (1.f + __expf(-xr));
  float z  = 1.f / (1.f + __expf(-xz));
  float nn = tanhf(gin + bi[2 * HID + hh] + r * (ghn + bh[2 * HID + hh]));

  float hold = h[idx];
  h[idx] = (1.f - z) * nn + z * hold;
}

// ---------------------------------------------------------------------------
// Final FC + ReLU: out[b][c] = relu(h[b][:] . Wfc[:,c] + bfc[c])  (f32, tiny)
// ---------------------------------------------------------------------------
__global__ __launch_bounds__(256) void fc_relu_kernel(
    const float* __restrict__ h, const float* __restrict__ W,
    const float* __restrict__ bias, float* __restrict__ out)
{
  int idx = blockIdx.x * 256 + threadIdx.x;
  if (idx >= BATCH * NCLS) return;
  int b = idx / NCLS;
  int c = idx - b * NCLS;
  float s = bias[c];
  const float* hr = h + (size_t)b * HID;
  for (int kk = 0; kk < HID; ++kk)
    s += hr[kk] * W[(size_t)kk * NCLS + c];
  out[idx] = s > 0.f ? s : 0.f;
}

// ---------------------------------------------------------------------------
extern "C" void kernel_launch(void* const* d_in, const int* in_sizes, int n_in,
                              void* d_out, int out_size, void* d_ws, size_t ws_size,
                              hipStream_t stream)
{
  (void)in_sizes; (void)n_in; (void)out_size; (void)ws_size;

  const float* x   = (const float*)d_in[0];   // [64,256,2048]
  const float* c0  = (const float*)d_in[1];   // [3,1,4,4,4]
  const float* c1  = (const float*)d_in[2];   // [3,4,8,4,4]
  const float* c2  = (const float*)d_in[3];   // [3,4,8,8,4]
  const float* c3  = (const float*)d_in[4];   // [3,4,8,8,1]
  const float* bi  = (const float*)d_in[5];   // [3,1024]
  const float* U   = (const float*)d_in[6];   // [1024,3072]
  const float* bh  = (const float*)d_in[7];   // [3,1024]
  const float* Wfc = (const float*)d_in[8];   // [1024,1000]
  const float* bfc = (const float*)d_in[9];   // [1000]
  // d_in[10] = length (constant 256, baked in)

  char* ws = (char*)d_ws;
  size_t off = 0;
  f16*   Weff = (f16*)(ws + off);   off += (size_t)DIN   * N3  * sizeof(f16);   // 12.6 MB
  f16*   U16  = (f16*)(ws + off);   off += (size_t)HID   * N3  * sizeof(f16);   //  6.3 MB
  f16*   GI   = (f16*)(ws + off);   off += (size_t)NROWS * N3  * sizeof(f16);   // 100.7 MB
  float* GH   = (float*)(ws + off); off += (size_t)BATCH * N3  * sizeof(float); //  0.8 MB
  float* h    = (float*)(ws + off); off += (size_t)BATCH * HID * sizeof(float); //  0.26 MB

  // Setup: TT -> dense operator, U -> f16, h = 0
  build_weff_kernel<<<(3 * DIN * HID + 255) / 256, 256, 0, stream>>>(c0, c1, c2, c3, Weff);
  f32_to_f16_kernel<<<(HID * N3 + 255) / 256, 256, 0, stream>>>(U, U16, HID * N3);
  zero_f32_kernel<<<(BATCH * HID + 255) / 256, 256, 0, stream>>>(h, BATCH * HID);

  // Phase 1: all input gates in one big WMMA GEMM: GI = X @ Weff  [16384 x 2048 x 3072]
  // 256 thr = 8 waves (4x2); wave tile 32x64 -> MF=2, NF=4 -> 8 WMMAs per k-step
  gemm_wmma_kernel<256, 128, 128, 4, 2, true>
      <<<dim3(N3 / 128, NROWS / 128), 256, 0, stream>>>(x, Weff, GI, NROWS, N3, DIN);

  // Phase 2: sequential recurrence, 2 launches per step (graph-captured)
  // 128 thr = 4 waves (2x2); wave tile 32x32 -> MF=NF=2; 48 WGs of parallelism
  for (int t = 0; t < T_STEPS; ++t) {
    gemm_wmma_kernel<128, 64, 64, 2, 2, false>
        <<<dim3(N3 / 64, 1), 128, 0, stream>>>(h, U16, GH, BATCH, N3, HID);
    gru_pointwise_kernel<<<(BATCH * HID) / 256, 256, 0, stream>>>(GI, GH, bi, bh, h, t);
  }

  // Final classifier
  fc_relu_kernel<<<(BATCH * NCLS + 255) / 256, 256, 0, stream>>>(h, Wfc, bfc, (float*)d_out);
}
